// SwinTransformerBlock_8254927143431
// MI455X (gfx1250) — compile-verified, matrix operand order checked by reading, not run on hardware
//
#include <hip/hip_runtime.h>
#include <hip/hip_bf16.h>

// ---------------- constants ----------------
constexpr int HD3   = 42;        // H = W = D
constexpr int CC    = 96;        // channels
constexpr int HEADS = 3;
constexpr int HDIM  = 32;        // head dim
constexpr int NWIN  = 216;       // 6*6*6 windows
constexpr int NTOK  = 343;       // 7*7*7 tokens per window
constexpr int NPAD  = 352;       // padded to 22*16
constexpr int C3    = 288;       // 3*C
constexpr int CM    = 384;       // mlp hidden
constexpr int TT    = 74088;     // 42^3 tokens
constexpr int TPAD  = 74112;     // padded to 16 (4632*16)
constexpr int SHIFT = 3;
constexpr float SCALE = 0.17677669529663687f;  // 32^-0.5

// workspace layout (bytes)
constexpr size_t OFF_WQKV  = 0;                                   // wqT  [C3][CC] f16
constexpr size_t OFF_WPROJ = OFF_WQKV + (size_t)CC * C3 * 2;      // wpT  [CC][CC]
constexpr size_t OFF_WFC1  = OFF_WPROJ + (size_t)CC * CC * 2;     // w1T  [CM][CC]
constexpr size_t OFF_WFC2  = OFF_WFC1 + (size_t)CC * CM * 2;      // w2T  [CC][CM]
constexpr size_t OFF_XW    = (size_t)1 << 20;
constexpr size_t SZ_XW     = (size_t)NWIN * NPAD * CC * 2;
constexpr size_t OFF_Q     = OFF_XW + SZ_XW;                      // q [win][h][n][hd]
constexpr size_t SZ_QH     = (size_t)NWIN * HEADS * NPAD * HDIM * 2;
constexpr size_t OFF_K     = OFF_Q + SZ_QH;                       // k [win][h][n][hd]
constexpr size_t OFF_V     = OFF_K + SZ_QH;                       // vT [win][h][hd][n]
constexpr size_t OFF_OW    = OFF_V + SZ_QH;
constexpr size_t OFF_OUT1  = OFF_OW + SZ_XW;
constexpr size_t OFF_Y2    = OFF_OUT1 + (size_t)TT * CC * 4;
constexpr size_t OFF_HH    = OFF_Y2 + (size_t)TPAD * CC * 2;

typedef __attribute__((ext_vector_type(16))) _Float16 v16h;
typedef __attribute__((ext_vector_type(8)))  _Float16 v8h;
typedef __attribute__((ext_vector_type(8)))  float    v8f;
typedef __attribute__((ext_vector_type(4)))  int      v4i;

// ---- optional gfx1250 async global->LDS path (guarded; falls back cleanly) ----
#if defined(__has_builtin)
# if __has_builtin(__builtin_amdgcn_global_load_async_to_lds_b128) && \
     __has_builtin(__builtin_amdgcn_s_wait_asynccnt)
#  define USE_ASYNC_LDS 1
# endif
#endif
#ifndef USE_ASYNC_LDS
# define USE_ASYNC_LDS 0
#endif

#if USE_ASYNC_LDS
typedef __attribute__((address_space(1))) v4i* gv4i_p;   // global int4*
typedef __attribute__((address_space(3))) v4i* lv4i_p;   // LDS int4*
__device__ __forceinline__ void async_copy_b128(void* lds, const void* g) {
  __builtin_amdgcn_global_load_async_to_lds_b128((gv4i_p)(void*)const_cast<void*>(g),
                                                 (lv4i_p)lds, 0, 0);
}
#endif

// ---------------- WMMA fragment helpers (CDNA5 16x16x32 f16 layouts) ----------------
// A operand (16x32, row-major source): lane&15 = row M; lanes 0-15 hold K {0-7,16-23},
// lanes 16-31 hold K {8-15,24-31}. With W^T storage this same loader also produces the
// B operand (lane&15 = column N of the original matrix).
__device__ __forceinline__ v16h load_frag_rowK(const _Float16* p, int ld) {
  int lane = threadIdx.x & 31;
  const _Float16* r = p + (size_t)(lane & 15) * ld + ((lane >> 4) << 3);
  v8h lo = *(const v8h*)(r);
  v8h hi = *(const v8h*)(r + 16);
  v16h f;
#pragma unroll
  for (int i = 0; i < 8; ++i) { f[i] = lo[i]; f[i + 8] = hi[i]; }
  return f;
}

__device__ __forceinline__ v8f wmma32(v16h a, v16h b, v8f c) {
  return __builtin_amdgcn_wmma_f32_16x16x32_f16(false, a, false, b, (short)0, c,
                                                false, false);
}

// ---------------- kernel 1: convert + transpose weights fp32 -> fp16 ----------------
__global__ void k_cvt_weights_t(const float* __restrict__ qkv_w, const float* __restrict__ proj_w,
                                const float* __restrict__ fc1_w, const float* __restrict__ fc2_w,
                                _Float16* wqT, _Float16* wpT, _Float16* w1T, _Float16* w2T) {
  int i = blockIdx.x * 256 + threadIdx.x;
  if (i < CC * C3) { int k = i / C3, n = i % C3; wqT[(size_t)n * CC + k] = (_Float16)qkv_w[i]; }
  if (i < CC * CC) { int k = i / CC, n = i % CC; wpT[(size_t)n * CC + k] = (_Float16)proj_w[i]; }
  if (i < CC * CM) { int k = i / CM, n = i % CM; w1T[(size_t)n * CC + k] = (_Float16)fc1_w[i]; }
  if (i < CM * CC) { int k = i / CC, n = i % CC; w2T[(size_t)n * CM + k] = (_Float16)fc2_w[i]; }
}

// ---------------- kernel 2: LN1 + cyclic shift + window partition -> f16 ----------------
__global__ void __launch_bounds__(128) k_ln1_window(const float* __restrict__ x,
                                                    const float* __restrict__ g,
                                                    const float* __restrict__ b,
                                                    _Float16* __restrict__ xw) {
  int wave = blockIdx.x * 4 + (threadIdx.x >> 5);
  int lane = threadIdx.x & 31;
  int win = wave / NPAD, n = wave % NPAD;
  _Float16* out = xw + ((size_t)win * NPAD + n) * CC;
  if (n >= NTOK) {  // zero padded rows
    out[lane] = (_Float16)0.f; out[lane + 32] = (_Float16)0.f; out[lane + 64] = (_Float16)0.f;
    return;
  }
  int hw = win / 36, ww = (win / 6) % 6, dw = win % 6;
  int hi = n / 49, wi = (n / 7) % 7, di = n % 7;
  int hs = (hw * 7 + hi + SHIFT) % HD3;
  int ws2 = (ww * 7 + wi + SHIFT) % HD3;
  int ds = (dw * 7 + di + SHIFT) % HD3;
  const float* xp = x + (((size_t)hs * HD3 + ws2) * HD3 + ds) * CC;
  float a0 = xp[lane], a1 = xp[lane + 32], a2 = xp[lane + 64];
  float s = a0 + a1 + a2;
#pragma unroll
  for (int o = 16; o; o >>= 1) s += __shfl_xor(s, o, 32);
  float mean = s * (1.f / 96.f);
  float d0 = a0 - mean, d1 = a1 - mean, d2 = a2 - mean;
  float vv = d0 * d0 + d1 * d1 + d2 * d2;
#pragma unroll
  for (int o = 16; o; o >>= 1) vv += __shfl_xor(vv, o, 32);
  float r = rsqrtf(vv * (1.f / 96.f) + 1e-5f);
  out[lane]      = (_Float16)(d0 * r * g[lane] + b[lane]);
  out[lane + 32] = (_Float16)(d1 * r * g[lane + 32] + b[lane + 32]);
  out[lane + 64] = (_Float16)(d2 * r * g[lane + 64] + b[lane + 64]);
}

// ---------------- kernel 3: QKV GEMM (per window, WMMA) ----------------
__global__ void __launch_bounds__(128) k_qkv(const _Float16* __restrict__ xw,
                                             const _Float16* __restrict__ wqT,
                                             const float* __restrict__ qkv_b,
                                             _Float16* __restrict__ qh,
                                             _Float16* __restrict__ kh,
                                             _Float16* __restrict__ vt) {
  int wave = blockIdx.x * 4 + (threadIdx.x >> 5);
  int lane = threadIdx.x & 31;
  int win = wave / 22, rt = wave % 22;
  const _Float16* A = xw + ((size_t)win * NPAD + rt * 16) * CC;
  v16h a0 = load_frag_rowK(A, CC);
  v16h a1 = load_frag_rowK(A + 32, CC);
  v16h a2 = load_frag_rowK(A + 64, CC);
  for (int nt = 0; nt < 18; ++nt) {
    const _Float16* B = wqT + (size_t)(nt * 16) * CC;
    if (nt + 1 < 18) __builtin_prefetch(wqT + (size_t)((nt + 1) * 16) * CC, 0, 0);
    v8f c = {};
    c = wmma32(a0, load_frag_rowK(B, CC), c);
    c = wmma32(a1, load_frag_rowK(B + 32, CC), c);
    c = wmma32(a2, load_frag_rowK(B + 64, CC), c);
    int col = nt * 16 + (lane & 15);
    float bias = qkv_b[col];
    int s3 = col / 96;
    int hh = (col % 96) / 32;
    int hd = col % 32;
    if (s3 == 2) {  // V stored transposed: [win][head][hd][token]
      _Float16* dst = vt + ((size_t)(win * HEADS + hh) * HDIM + hd) * NPAD;
#pragma unroll
      for (int i = 0; i < 8; ++i) {
        int row = rt * 16 + i + ((lane >> 4) << 3);
        dst[row] = (_Float16)(c[i] + bias);
      }
    } else {
      _Float16* dst = (s3 == 0 ? qh : kh) + ((size_t)(win * HEADS + hh) * NPAD) * HDIM + hd;
      float scl = (s3 == 0) ? SCALE : 1.f;
#pragma unroll
      for (int i = 0; i < 8; ++i) {
        int row = rt * 16 + i + ((lane >> 4) << 3);
        dst[(size_t)row * HDIM] = (_Float16)((c[i] + bias) * scl);
      }
    }
  }
}

// ---------------- kernel 4: windowed attention (scores + softmax + P.V) ----------------
// one wave per block; V^T tile staged into LDS (async on gfx1250 toolchains that expose it)
__global__ void __launch_bounds__(32) k_attn(const _Float16* __restrict__ qh,
                                             const _Float16* __restrict__ kh,
                                             const _Float16* __restrict__ vt,
                                             const float* __restrict__ rpb,
                                             const float* __restrict__ mask,
                                             _Float16* __restrict__ ow) {
  __shared__ __align__(16) float    Ssc[16 * NPAD];     // 22528 B score strip (f32)
  __shared__ __align__(16) _Float16 Pst[16 * NPAD];     // 11264 B prob strip (f16)
  __shared__ __align__(16) _Float16 Vl[HDIM * NPAD];    // 22528 B V^T tile
  int lane = threadIdx.x & 31;
  int wh = blockIdx.y;
  int win = wh / HEADS, head = wh % HEADS;
  int rb = blockIdx.x * 16;                             // query row base
  size_t base = (size_t)(win * HEADS + head) * NPAD * HDIM;
  const _Float16* vsrc = vt + base;                     // [HDIM][NPAD]

  // stage V^T tile into LDS (overlaps with score/softmax phase when async)
#if USE_ASYNC_LDS
  for (int t = lane; t < (HDIM * NPAD) / 8; t += 32)
    async_copy_b128(Vl + (size_t)t * 8, vsrc + (size_t)t * 8);
#else
  for (int t = lane; t < (HDIM * NPAD) / 8; t += 32)
    *(v8h*)(Vl + (size_t)t * 8) = *(const v8h*)(vsrc + (size_t)t * 8);
#endif

  v16h aq = load_frag_rowK(qh + base + (size_t)rb * HDIM, HDIM);  // 16x32 Q strip

  // score strip: S = Q.K^T + bias + mask   (one wmma per 16x16 tile; Kdim=32)
  for (int jt = 0; jt < 22; ++jt) {
    v16h bk = load_frag_rowK(kh + base + (size_t)jt * 16 * HDIM, HDIM);  // K^T operand
    v8f c = {};
    c = wmma32(aq, bk, c);
    int m = jt * 16 + (lane & 15);
#pragma unroll
    for (int i = 0; i < 8; ++i) {
      int rl = i + ((lane >> 4) << 3);
      int n = rb + rl;
      float sv = c[i];
      if (n < NTOK && m < NTOK) {
        int dn0 = n / 49 - m / 49;
        int dn1 = (n / 7) % 7 - (m / 7) % 7;
        int dn2 = n % 7 - m % 7;
        int ridx = (dn0 + 6) * 169 + (dn1 + 6) * 13 + (dn2 + 6);
        sv += rpb[ridx * HEADS + head] + mask[((size_t)win * NTOK + n) * NTOK + m];
      }
      Ssc[rl * NPAD + m] = sv;
    }
  }
  __syncthreads();

  // softmax over m (2 lanes per row); write normalized probs as f16 into Pst
  {
    int rl = lane >> 1, par = lane & 1;
    float* row = Ssc + rl * NPAD;
    _Float16* prow = Pst + rl * NPAD;
    float mx = -3.4e38f;
    for (int m = par; m < NTOK; m += 2) mx = fmaxf(mx, row[m]);
    mx = fmaxf(mx, __shfl_xor(mx, 1, 32));
    float sum = 0.f;
    for (int m = par; m < NTOK; m += 2) {
      float e = __expf(row[m] - mx);
      row[m] = e;
      sum += e;
    }
    sum += __shfl_xor(sum, 1, 32);
    float inv = 1.f / sum;
    for (int m = par; m < NTOK; m += 2) prow[m] = (_Float16)(row[m] * inv);
    for (int m = NTOK + par; m < NPAD; m += 2) prow[m] = (_Float16)0.f;  // padded cols
  }
#if USE_ASYNC_LDS
  __builtin_amdgcn_s_wait_asynccnt(0);
#endif
  __syncthreads();

  // O = P.V : A = f16 prob strip from LDS, B = V^T rows from LDS (both b128 loads)
  for (int nt = 0; nt < 2; ++nt) {
    v8f acc = {};
    for (int kt = 0; kt < 11; ++kt) {
      v16h ap = load_frag_rowK(Pst + kt * 32, NPAD);
      v16h bv = load_frag_rowK(Vl + (size_t)(nt * 16) * NPAD + kt * 32, NPAD);
      acc = wmma32(ap, bv, acc);
    }
    int col = head * HDIM + nt * 16 + (lane & 15);
#pragma unroll
    for (int i = 0; i < 8; ++i) {
      int n = rb + i + ((lane >> 4) << 3);
      if (n < NTOK) ow[((size_t)win * NPAD + n) * CC + col] = (_Float16)acc[i];
    }
  }
}

// ---------------- kernel 5: proj GEMM + window reverse + unshift + residual ----------------
__global__ void __launch_bounds__(128) k_proj(const _Float16* __restrict__ ow,
                                              const _Float16* __restrict__ wpT,
                                              const float* __restrict__ proj_b,
                                              const float* __restrict__ x,
                                              float* __restrict__ out1) {
  int wave = blockIdx.x * 4 + (threadIdx.x >> 5);
  int lane = threadIdx.x & 31;
  int win = wave / 22, rt = wave % 22;
  const _Float16* A = ow + ((size_t)win * NPAD + rt * 16) * CC;
  v16h a0 = load_frag_rowK(A, CC);
  v16h a1 = load_frag_rowK(A + 32, CC);
  v16h a2 = load_frag_rowK(A + 64, CC);
  int hw = win / 36, ww = (win / 6) % 6, dw = win % 6;
  for (int nt = 0; nt < 6; ++nt) {
    const _Float16* B = wpT + (size_t)(nt * 16) * CC;
    v8f c = {};
    c = wmma32(a0, load_frag_rowK(B, CC), c);
    c = wmma32(a1, load_frag_rowK(B + 32, CC), c);
    c = wmma32(a2, load_frag_rowK(B + 64, CC), c);
    int col = nt * 16 + (lane & 15);
    float pb = proj_b[col];
#pragma unroll
    for (int i = 0; i < 8; ++i) {
      int n = rt * 16 + i + ((lane >> 4) << 3);
      if (n < NTOK) {
        int hi = n / 49, wi = (n / 7) % 7, di = n % 7;
        int hs = (hw * 7 + hi + SHIFT) % HD3;
        int ws2 = (ww * 7 + wi + SHIFT) % HD3;
        int ds = (dw * 7 + di + SHIFT) % HD3;
        size_t sp = (((size_t)hs * HD3 + ws2) * HD3 + ds) * CC + col;
        out1[sp] = x[sp] + c[i] + pb;
      }
    }
  }
}

// ---------------- kernel 6: LN2 -> f16 (token-major, padded) ----------------
__global__ void __launch_bounds__(128) k_ln2(const float* __restrict__ out1,
                                             const float* __restrict__ g,
                                             const float* __restrict__ b,
                                             _Float16* __restrict__ y2) {
  int wave = blockIdx.x * 4 + (threadIdx.x >> 5);
  int lane = threadIdx.x & 31;
  _Float16* o = y2 + (size_t)wave * CC;
  if (wave >= TT) {
    o[lane] = (_Float16)0.f; o[lane + 32] = (_Float16)0.f; o[lane + 64] = (_Float16)0.f;
    return;
  }
  const float* xp = out1 + (size_t)wave * CC;
  float a0 = xp[lane], a1 = xp[lane + 32], a2 = xp[lane + 64];
  float s = a0 + a1 + a2;
#pragma unroll
  for (int off = 16; off; off >>= 1) s += __shfl_xor(s, off, 32);
  float mean = s * (1.f / 96.f);
  float d0 = a0 - mean, d1 = a1 - mean, d2 = a2 - mean;
  float vv = d0 * d0 + d1 * d1 + d2 * d2;
#pragma unroll
  for (int off = 16; off; off >>= 1) vv += __shfl_xor(vv, off, 32);
  float r = rsqrtf(vv * (1.f / 96.f) + 1e-5f);
  o[lane]      = (_Float16)(d0 * r * g[lane] + b[lane]);
  o[lane + 32] = (_Float16)(d1 * r * g[lane + 32] + b[lane + 32]);
  o[lane + 64] = (_Float16)(d2 * r * g[lane + 64] + b[lane + 64]);
}

// ---------------- kernel 7: fc1 GEMM + exact GELU -> f16 ----------------
__global__ void __launch_bounds__(128) k_fc1(const _Float16* __restrict__ y2,
                                             const _Float16* __restrict__ w1T,
                                             const float* __restrict__ fc1_b,
                                             _Float16* __restrict__ hh) {
  int wid = threadIdx.x >> 5, lane = threadIdx.x & 31;
  int rt = blockIdx.x;  // 16-row strip
  const _Float16* A = y2 + (size_t)rt * 16 * CC;
  v16h a0 = load_frag_rowK(A, CC);
  v16h a1 = load_frag_rowK(A + 32, CC);
  v16h a2 = load_frag_rowK(A + 64, CC);
  for (int t = 0; t < 6; ++t) {
    int nt = wid * 6 + t;   // 4 waves cover 24 n-tiles (384 cols)
    const _Float16* B = w1T + (size_t)(nt * 16) * CC;
    v8f c = {};
    c = wmma32(a0, load_frag_rowK(B, CC), c);
    c = wmma32(a1, load_frag_rowK(B + 32, CC), c);
    c = wmma32(a2, load_frag_rowK(B + 64, CC), c);
    int col = nt * 16 + (lane & 15);
    float b1 = fc1_b[col];
#pragma unroll
    for (int i = 0; i < 8; ++i) {
      int row = rt * 16 + i + ((lane >> 4) << 3);
      float u = c[i] + b1;
      float gel = 0.5f * u * (1.f + erff(u * 0.70710678118654752f));
      hh[(size_t)row * CM + col] = (_Float16)gel;
    }
  }
}

// ---------------- kernel 8: fc2 GEMM + residual -> out ----------------
__global__ void __launch_bounds__(128) k_fc2(const _Float16* __restrict__ hh,
                                             const _Float16* __restrict__ w2T,
                                             const float* __restrict__ fc2_b,
                                             const float* __restrict__ out1,
                                             float* __restrict__ out) {
  int wave = blockIdx.x * 4 + (threadIdx.x >> 5);   // 16-row strip id
  int lane = threadIdx.x & 31;
  const _Float16* A = hh + (size_t)wave * 16 * CM;
  v8f acc[6] = {};
  for (int kt = 0; kt < 12; ++kt) {
    v16h a = load_frag_rowK(A + kt * 32, CM);
    if (kt + 1 < 12) __builtin_prefetch(A + (kt + 1) * 32, 0, 0);
#pragma unroll
    for (int nt = 0; nt < 6; ++nt) {
      v16h b = load_frag_rowK(w2T + (size_t)(nt * 16) * CM + kt * 32, CM);
      acc[nt] = wmma32(a, b, acc[nt]);
    }
  }
#pragma unroll
  for (int nt = 0; nt < 6; ++nt) {
    int col = nt * 16 + (lane & 15);
    float b2 = fc2_b[col];
#pragma unroll
    for (int i = 0; i < 8; ++i) {
      int row = wave * 16 + i + ((lane >> 4) << 3);
      if (row < TT) {
        size_t idx = (size_t)row * CC + col;
        out[idx] = out1[idx] + acc[nt][i] + b2;
      }
    }
  }
}

// ---------------- host launcher ----------------
extern "C" void kernel_launch(void* const* d_in, const int* in_sizes, int n_in,
                              void* d_out, int out_size, void* d_ws, size_t ws_size,
                              hipStream_t stream) {
  (void)in_sizes; (void)n_in; (void)out_size; (void)ws_size;
  const float* x      = (const float*)d_in[0];
  const float* mask   = (const float*)d_in[1];
  const float* ln1_g  = (const float*)d_in[2];
  const float* ln1_b  = (const float*)d_in[3];
  const float* qkv_w  = (const float*)d_in[4];
  const float* qkv_b  = (const float*)d_in[5];
  const float* rpb    = (const float*)d_in[6];
  const float* proj_w = (const float*)d_in[7];
  const float* proj_b = (const float*)d_in[8];
  const float* ln2_g  = (const float*)d_in[9];
  const float* ln2_b  = (const float*)d_in[10];
  const float* fc1_w  = (const float*)d_in[11];
  const float* fc1_b  = (const float*)d_in[12];
  const float* fc2_w  = (const float*)d_in[13];
  const float* fc2_b  = (const float*)d_in[14];
  float* out = (float*)d_out;
  char* ws = (char*)d_ws;

  _Float16* wqT = (_Float16*)(ws + OFF_WQKV);
  _Float16* wpT = (_Float16*)(ws + OFF_WPROJ);
  _Float16* w1T = (_Float16*)(ws + OFF_WFC1);
  _Float16* w2T = (_Float16*)(ws + OFF_WFC2);
  _Float16* xw  = (_Float16*)(ws + OFF_XW);
  _Float16* qh  = (_Float16*)(ws + OFF_Q);
  _Float16* kh  = (_Float16*)(ws + OFF_K);
  _Float16* vt  = (_Float16*)(ws + OFF_V);
  _Float16* ow  = (_Float16*)(ws + OFF_OW);
  float*    o1  = (float*)(ws + OFF_OUT1);
  _Float16* y2  = (_Float16*)(ws + OFF_Y2);
  _Float16* hh  = (_Float16*)(ws + OFF_HH);

  k_cvt_weights_t<<<144, 256, 0, stream>>>(qkv_w, proj_w, fc1_w, fc2_w, wqT, wpT, w1T, w2T);
  k_ln1_window<<<(NWIN * NPAD) / 4, 128, 0, stream>>>(x, ln1_g, ln1_b, xw);
  k_qkv<<<(NWIN * 22) / 4, 128, 0, stream>>>(xw, wqT, qkv_b, qh, kh, vt);
  k_attn<<<dim3(NPAD / 16, NWIN * HEADS), 32, 0, stream>>>(qh, kh, vt, rpb, mask, ow);
  k_proj<<<(NWIN * 22) / 4, 128, 0, stream>>>(ow, wpT, proj_b, x, o1);
  k_ln2<<<TPAD / 4, 128, 0, stream>>>(o1, ln2_g, ln2_b, y2);
  k_fc1<<<TPAD / 16, 128, 0, stream>>>(y2, w1T, fc1_b, hh);
  k_fc2<<<TPAD / 64, 128, 0, stream>>>(hh, w2T, fc2_b, o1, out);
}
